// GraphLSTM_73735998538262
// MI455X (gfx1250) — matrix operand order repaired and checked by reading, not compiled, not run on hardware
//
#include <hip/hip_runtime.h>
#include <hip/hip_bf16.h>

#define NUM_TREES 64
#define DEPTH     11
#define NPT       2047          // nodes per tree
#define NTOT      (NUM_TREES * NPT)   // 131008 = 16 * 8188
#define HD        256
#define IOUC      768
#define WCATC     1024

typedef __attribute__((ext_vector_type(16))) __bf16 bf16x16;
typedef __attribute__((ext_vector_type(8)))  __bf16 bf16x8;
typedef __attribute__((ext_vector_type(8)))  float  v8f;

__device__ __forceinline__ float sigm(float x) { return 1.0f / (1.0f + __expf(-x)); }

// Async copy of 16 bytes global -> LDS (gfx1250 ASYNCcnt path).
// dsaddr = low 32 bits of the flat shared pointer (addr[31:0] == LDS offset).
__device__ __forceinline__ void async_copy_b128(void* lds_dst, const void* gsrc) {
    unsigned ldsoff = (unsigned)(size_t)lds_dst;
    asm volatile("global_load_async_to_lds_b128 %0, %1, off"
                 :: "v"(ldsoff), "v"(gsrc) : "memory");
}
__device__ __forceinline__ void wait_asynccnt0() {
    asm volatile("s_wait_asynccnt 0" ::: "memory");
}

// ---------------------------------------------------------------------------
// K0: convert weights to bf16. wiou_b = W_iou [768][256];
//     wcat_b rows 0..255 = U_f_w, rows 256..1023 = U_iou  ([1024][256]).
// ---------------------------------------------------------------------------
__global__ void prep_weights(const float* __restrict__ Wiou,
                             const float* __restrict__ Uiou,
                             const float* __restrict__ Ufw,
                             __bf16* __restrict__ wiou_b,
                             __bf16* __restrict__ wcat_b) {
    int i = blockIdx.x * blockDim.x + threadIdx.x;   // covers 1024*256
    if (i < IOUC * HD) wiou_b[i] = (__bf16)Wiou[i];
    if (i < WCATC * HD) {
        int r = i >> 8, k = i & 255;
        float v = (r < HD) ? Ufw[r * HD + k] : Uiou[(r - HD) * HD + k];
        wcat_b[i] = (__bf16)v;
    }
}

// ---------------------------------------------------------------------------
// WMMA fragment helpers (CDNA5 16x16x32 bf16 layouts)
// ---------------------------------------------------------------------------
__device__ __forceinline__ bf16x16 load_a_frag(const __bf16* As /*16x256 row-major LDS*/,
                                               int lane, int k0) {
    int row  = lane & 15;
    int half = (lane >> 4) & 1;
    int ka   = k0 + half * 16;
    bf16x8 lo = *(const bf16x8*)&As[row * HD + ka];
    bf16x8 hi = *(const bf16x8*)&As[row * HD + ka + 8];
    return __builtin_shufflevector(lo, hi, 0,1,2,3,4,5,6,7,8,9,10,11,12,13,14,15);
}

__device__ __forceinline__ bf16x16 load_b_frag(const __bf16* __restrict__ Wrow /*=W + jcol*256*/,
                                               int lane, int k0) {
    int half = (lane >> 4) & 1;
    int kb   = k0 + half * 16;
    return *(const bf16x16*)(Wrow + kb);  // 32B contiguous per lane
}

// ---------------------------------------------------------------------------
// K1: iou_all[n][0..767] = emb[features[n]] @ W_iou.T  (all N nodes)
// One block per 16-node tile; 8 waves x 6 col-tiles; A frags held in VGPRs.
// ---------------------------------------------------------------------------
__global__ __launch_bounds__(256) void iou_gemm(const int*   __restrict__ feats,
                                                const float* __restrict__ emb,
                                                const __bf16* __restrict__ Wb,
                                                float* __restrict__ iou_all) {
    __shared__ __bf16 As[16 * HD];   // 8 KB
    int tid = threadIdx.x;
    int rt  = blockIdx.x;
    // stage gathered embeddings as bf16 (conversion is inherent here)
    int row  = tid >> 4;
    int cg   = (tid & 15) << 4;
    int node = rt * 16 + row;
    const float* er = emb + (size_t)feats[node] * HD + cg;
    #pragma unroll
    for (int e = 0; e < 16; ++e) As[row * HD + cg + e] = (__bf16)er[e];
    __syncthreads();

    int wave = tid >> 5, lane = tid & 31;
    int half = (lane >> 4) & 1, ncol = lane & 15;

    bf16x16 afr[8];                       // hoist: whole K=256 of the A tile
    #pragma unroll
    for (int ks = 0; ks < 8; ++ks) afr[ks] = load_a_frag(As, lane, ks * 32);

    for (int q = 0; q < 6; ++q) {
        int ct = wave + 8 * q;                 // 0..47
        int jcol = ct * 16 + ncol;             // output column
        const __bf16* bptr = Wb + (size_t)jcol * HD;
        v8f acc = {};
        #pragma unroll
        for (int ks = 0; ks < 8; ++ks) {
            bf16x16 b = load_b_frag(bptr, lane, ks * 32);
            acc = __builtin_amdgcn_wmma_f32_16x16x32_bf16(false, afr[ks], false, b,
                                                          (short)0, acc, false, false);
        }
        #pragma unroll
        for (int r = 0; r < 8; ++r) {
            int m = r + half * 8;
            iou_all[(size_t)(rt * 16 + m) * IOUC + jcol] = acc[r];
        }
    }
}

// ---------------------------------------------------------------------------
// K2: leaf gates: c = sig(i)*tanh(u); h = sig(o)*tanh(c). Writes f32 h/c and
// a bf16 mirror of h for the next level's async A staging.
// ---------------------------------------------------------------------------
__global__ void leaf_gates(const float* __restrict__ iou_all,
                           const float* __restrict__ b_iou,
                           float* __restrict__ h, float* __restrict__ c,
                           __bf16* __restrict__ hb) {
    int idx  = blockIdx.x * blockDim.x + threadIdx.x;  // 65536*256 exactly
    int li   = idx >> 8, j = idx & 255;
    int tree = li >> 10, q = li & 1023;
    int node = tree * NPT + 1023 + q;
    const float* ir = iou_all + (size_t)node * IOUC;
    float iv = ir[j]       + b_iou[j];
    float ov = ir[256 + j] + b_iou[256 + j];
    float uv = ir[512 + j] + b_iou[512 + j];
    float cn = sigm(iv) * tanhf(uv);
    float hn = sigm(ov) * tanhf(cn);
    h [(size_t)node * HD + j] = hn;
    c [(size_t)node * HD + j] = cn;
    hb[(size_t)node * HD + j] = (__bf16)hn;
}

// ---------------------------------------------------------------------------
// K3: one internal level. Block = 8 parents -> 16 children (one A tile).
// Child h staged via global_load_async_to_lds_b128 from the bf16 h mirror.
// ---------------------------------------------------------------------------
__global__ __launch_bounds__(256) void level_kernel(int l,
                                                    const __bf16* __restrict__ Wcat,
                                                    const float* __restrict__ iou_all,
                                                    const float* __restrict__ b_iou,
                                                    const float* __restrict__ ufb,
                                                    float* __restrict__ h,
                                                    float* __restrict__ c,
                                                    __bf16* __restrict__ hb) {
    __shared__ __bf16 As[16 * HD];     //  8 KB child h (bf16)
    __shared__ float  cinS[8 * HD];    //  8 KB c_in per parent
    __shared__ float  iouS[8 * IOUC];  // 24 KB iou_n per parent
    int tid   = threadIdx.x;
    int pbase = blockIdx.x * 8;        // flat parent index base
    int lmask = (1 << l) - 1;

    // stage 16 child h rows (512B each) via async copy: 256 thr x 32B
    {
        int row = tid >> 4;            // child row 0..15
        int seg = (tid & 15) << 5;     // 32B segment within 512B row
        int pi  = pbase + (row >> 1);
        int tree = pi >> l;
        int local = lmask + (pi & lmask);
        int cnode = tree * NPT + 2 * local + 1 + (row & 1);
        const char* src = (const char*)(hb + (size_t)cnode * HD) + seg;
        char*       dst = (char*)As + row * (HD * 2) + seg;
        async_copy_b128(dst,      src);
        async_copy_b128(dst + 16, src + 16);
        wait_asynccnt0();
    }
    __syncthreads();

    int wave = tid >> 5, lane = tid & 31;
    int half = (lane >> 4) & 1, ncol = lane & 15;

    bf16x16 afr[8];                    // hoist A fragments for all K
    #pragma unroll
    for (int ks = 0; ks < 8; ++ks) afr[ks] = load_a_frag(As, lane, ks * 32);

    for (int q = 0; q < 8; ++q) {
        int ct   = wave + 8 * q;              // 0..63 over 1024 cols
        int jcol = ct * 16 + ncol;
        const __bf16* bptr = Wcat + (size_t)jcol * HD;
        v8f acc = {};
        #pragma unroll
        for (int ks = 0; ks < 8; ++ks) {
            bf16x16 b = load_b_frag(bptr, lane, ks * 32);
            acc = __builtin_amdgcn_wmma_f32_16x16x32_bf16(false, afr[ks], false, b,
                                                          (short)0, acc, false, false);
        }
        if (ct < 16) {
            // forget-gate columns: f = sig(acc + U_f_b); c_in = fL*cL + fR*cR
            int hc = jcol;                    // 0..255
            float bias = ufb[hc];
            #pragma unroll
            for (int k2 = 0; k2 < 4; ++k2) {
                int p   = k2 + half * 4;      // local parent 0..7
                int ppi = pbase + p;
                int t2  = ppi >> l;
                int cL  = t2 * NPT + 2 * (lmask + (ppi & lmask)) + 1;
                float fL = sigm(acc[2 * k2]     + bias);
                float fR = sigm(acc[2 * k2 + 1] + bias);
                float cLv = c[(size_t)cL * HD + hc];
                float cRv = c[(size_t)(cL + 1) * HD + hc];
                cinS[p * HD + hc] = fL * cLv + fR * cRv;
            }
        } else {
            // iou columns: pair-sum children + iou_all + b_iou
            int jc = jcol - 256;              // 0..767
            #pragma unroll
            for (int k2 = 0; k2 < 4; ++k2) {
                int p   = k2 + half * 4;
                int ppi = pbase + p;
                int t2  = ppi >> l;
                int pnode = t2 * NPT + lmask + (ppi & lmask);
                float s = acc[2 * k2] + acc[2 * k2 + 1];
                iouS[p * IOUC + jc] = iou_all[(size_t)pnode * IOUC + jc] + s + b_iou[jc];
            }
        }
    }
    __syncthreads();

    // gates for 8 parents x 256 hidden
    #pragma unroll
    for (int e = 0; e < 8; ++e) {
        int idx = tid + 256 * e;              // 0..2047
        int p = idx >> 8, hc = idx & 255;
        int ppi = pbase + p;
        int t2  = ppi >> l;
        int pnode = t2 * NPT + lmask + (ppi & lmask);
        float iv = iouS[p * IOUC + hc];
        float ov = iouS[p * IOUC + 256 + hc];
        float uv = iouS[p * IOUC + 512 + hc];
        float cn = sigm(iv) * tanhf(uv) + cinS[p * HD + hc];
        float hn = sigm(ov) * tanhf(cn);
        h [(size_t)pnode * HD + hc] = hn;
        c [(size_t)pnode * HD + hc] = cn;
        hb[(size_t)pnode * HD + hc] = (__bf16)hn;
    }
}

// ---------------------------------------------------------------------------
// K4: per-tree mean pool + linear head (tiny, fp32 scalar)
// ---------------------------------------------------------------------------
__global__ __launch_bounds__(256) void final_kernel(const float* __restrict__ h,
                                                    const float* __restrict__ lw,
                                                    const float* __restrict__ lb,
                                                    float* __restrict__ out) {
    __shared__ float hg[HD];
    int tree = blockIdx.x, tid = threadIdx.x;
    const float* hb = h + (size_t)tree * NPT * HD;
    float s = 0.0f;
    for (int n = 0; n < NPT; ++n) s += hb[(size_t)n * HD + tid];
    hg[tid] = s * (1.0f / (float)NPT);
    __syncthreads();
    if (tid < 104) {
        float acc = lb[tid];
        const float* wr = lw + tid * HD;
        #pragma unroll 4
        for (int k = 0; k < HD; ++k) acc += hg[k] * wr[k];
        out[tree * 104 + tid] = acc;
    }
}

// ---------------------------------------------------------------------------
extern "C" void kernel_launch(void* const* d_in, const int* in_sizes, int n_in,
                              void* d_out, int out_size, void* d_ws, size_t ws_size,
                              hipStream_t stream) {
    (void)in_sizes; (void)n_in; (void)out_size; (void)ws_size;
    const int*   feats = (const int*)d_in[0];
    const float* emb   = (const float*)d_in[1];
    const float* Wiou  = (const float*)d_in[2];
    const float* Uiou  = (const float*)d_in[3];
    const float* biou  = (const float*)d_in[4];
    const float* Ufw   = (const float*)d_in[5];
    const float* Ufb   = (const float*)d_in[6];
    const float* lw    = (const float*)d_in[7];
    const float* lb    = (const float*)d_in[8];
    float* out = (float*)d_out;

    char* ws = (char*)d_ws;
    size_t o = 0;
    __bf16* wiou_b = (__bf16*)(ws + o); o += (size_t)IOUC  * HD * 2;
    __bf16* wcat_b = (__bf16*)(ws + o); o += (size_t)WCATC * HD * 2;
    o = (o + 255) & ~(size_t)255;
    float*  hbuf = (float*) (ws + o); o += (size_t)NTOT * HD * 4;
    float*  cbuf = (float*) (ws + o); o += (size_t)NTOT * HD * 4;
    float*  ioub = (float*) (ws + o); o += (size_t)NTOT * IOUC * 4;
    __bf16* hbf  = (__bf16*)(ws + o); o += (size_t)NTOT * HD * 2;

    prep_weights<<<(WCATC * HD + 255) / 256, 256, 0, stream>>>(Wiou, Uiou, Ufw, wiou_b, wcat_b);
    iou_gemm<<<NTOT / 16, 256, 0, stream>>>(feats, emb, wiou_b, ioub);
    leaf_gates<<<(NUM_TREES * 1024 * HD) / 256, 256, 0, stream>>>(ioub, biou, hbuf, cbuf, hbf);
    for (int l = DEPTH - 2; l >= 0; --l) {
        int blocks = (NUM_TREES << l) / 8;    // 8 parents per block; always exact
        level_kernel<<<blocks, 256, 0, stream>>>(l, wcat_b, ioub, biou, Ufb, hbuf, cbuf, hbf);
    }
    final_kernel<<<NUM_TREES, 256, 0, stream>>>(hbuf, lw, lb, out);
}
